// SimpleMambaPolicy_86208583565492
// MI455X (gfx1250) — matrix operand order repaired and checked by reading, not compiled, not run on hardware
//
#include <hip/hip_runtime.h>

// ---------------------------------------------------------------------------
// SimpleMambaPolicy on MI455X (gfx1250, wave32, WMMA + async LDS staging)
//   GEMM1 (bf16 wmma) -> per-hidden decay scan (chunked, 3-phase) -> tanh
//   -> GEMM2 (bf16 wmma) -> Gaussian log-prob reduction
// ---------------------------------------------------------------------------

#define L_  2048
#define B_  32
#define D_  256
#define H_  1024
#define A_  64
#define CL_ 16
#define NC_ (L_ / CL_)   // 128 chunks

typedef __bf16 bf16_t;
typedef bf16_t v16bf __attribute__((ext_vector_type(16)));
typedef float  v8f   __attribute__((ext_vector_type(8)));
typedef unsigned int   u32;
typedef unsigned short u16;
typedef u32 v4u __attribute__((ext_vector_type(4)));

struct Frag32 { v4u lo, hi; };   // 32 bytes == v16bf

static __device__ __forceinline__ u16 f2bf(float x) {
  u32 u = __float_as_uint(x);
  u32 r = u + 0x7FFFu + ((u >> 16) & 1u);   // round-to-nearest-even
  return (u16)(r >> 16);
}

// Fast tanh via hardware transcendentals: v_exp_f32 + v_rcp_f32.
// tanh(x) = 1 - 2/(exp(2x)+1); saturates correctly at +/-inf.
static __device__ __forceinline__ float fast_tanh(float x) {
  float e = __expf(2.f * x);
  return 1.f - 2.f * __builtin_amdgcn_rcpf(e + 1.f);
}

// A fragment: 16x32 bf16. Lane<16 holds K {k0..k0+7, k0+16..k0+23} of row
// row0+(lane&15); lane>=16 holds K {+8..+15, +24..+31}. (ISA 7.12.2)
static __device__ __forceinline__ v16bf load_a_frag(const u16* base, int row0,
                                                    int k0, int ld, int lane) {
  int r  = row0 + (lane & 15);
  int kb = k0 + ((lane & 16) ? 8 : 0);
  const u16* p = base + (size_t)r * ld + kb;
  Frag32 f;
  f.lo = *(const v4u*)(p);
  f.hi = *(const v4u*)(p + 16);
  return __builtin_bit_cast(v16bf, f);
}

// B fragment: 32x16 bf16 (KxN). W stored [N][K] row-major, so column n of the
// KxN operand is contiguous bf16. Lane<16: K {k0..k0+7, k0+16..k0+23}; lane>=16: {+8..+15, +24..+31}.
static __device__ __forceinline__ v16bf load_b_frag(const u16* W, int n0,
                                                    int k0, int ld, int lane) {
  int n  = n0 + (lane & 15);
  int kb = k0 + ((lane & 16) ? 8 : 0);
  const u16* p = W + (size_t)n * ld + kb;
  Frag32 f;
  f.lo = *(const v4u*)(p);
  f.hi = *(const v4u*)(p + 16);
  return __builtin_bit_cast(v16bf, f);
}

static __device__ __forceinline__ v8f wmma_bf16(v16bf a, v16bf b, v8f c) {
  return __builtin_amdgcn_wmma_f32_16x16x32_bf16(false, a, false, b,
                                                 (short)0, c, false, false);
}

// Async global -> LDS copy (16 B per lane), tracked by ASYNCcnt.
// lds_off is the byte offset within the workgroup's LDS allocation
// (hardware adds LDS_BASE; dynamic-shared region starts at offset 0).
static __device__ __forceinline__ void async_ld_b128(u32 lds_off, const void* g) {
  asm volatile("global_load_async_to_lds_b128 %0, %1, off"
               :: "v"(lds_off), "v"((unsigned long long)(uintptr_t)g)
               : "memory");
}
static __device__ __forceinline__ void wait_async0() {
  asm volatile("s_wait_asynccnt 0x0" ::: "memory");
}

// ---------------------------------------------------------------------------
// Prep: f32 -> bf16 conversion (states, B_w, C_w)
// ---------------------------------------------------------------------------
__global__ void __launch_bounds__(256)
cvt_bf16_kernel(const float* __restrict__ src, u16* __restrict__ dst, int n) {
  int i = blockIdx.x * blockDim.x + threadIdx.x;
  int stride = gridDim.x * blockDim.x;
  for (; i < n; i += stride) dst[i] = f2bf(src[i]);
}

// ---------------------------------------------------------------------------
// Pass 1: per (chunk, 64-hidden-col group) compute chunk-final local state
//   E_c = sum_{s in chunk} decay^{end-s} * (states_s @ Bw^T + B_b)
// B_w fragments live in VGPRs across the whole chunk; A fragments for a
// timestep are batch-loaded so loads clause + overlap the WMMA pipeline.
// ---------------------------------------------------------------------------
__global__ void __launch_bounds__(128)
chunk_state_kernel(const u16* __restrict__ statesB, const u16* __restrict__ BwB,
                   const float* __restrict__ A_log, const float* __restrict__ B_b,
                   float* __restrict__ E) {
  const int c    = blockIdx.x;
  const int hg   = blockIdx.y;
  const int tid  = threadIdx.x;
  const int w    = tid >> 5;
  const int lane = tid & 31;
  const int n0   = hg * 64 + w * 16;
  const int ncol = n0 + (lane & 15);
  const float dec = __expf(A_log[ncol]);
  const float bb  = B_b[ncol];

  // Loop-invariant B_w fragments (8 K-steps) held in VGPRs
  v16bf fbw[8];
#pragma unroll
  for (int k = 0; k < 8; ++k) fbw[k] = load_b_frag(BwB, n0, k * 32, D_, lane);

  v8f e0 = {}; v8f e1 = {};
  for (int t = 0; t < CL_; ++t) {
    const int l = c * CL_ + t;
    if (t + 1 < CL_)   // hint next timestep's states tile into near caches
      __builtin_prefetch(statesB + (size_t)(l + 1) * B_ * D_ + lane * 16, 0, 3);

    // Batch all 16 A fragments for this timestep, then run the WMMA chain.
    v16bf fa0[8], fa1[8];
#pragma unroll
    for (int k = 0; k < 8; ++k) {
      fa0[k] = load_a_frag(statesB, l * B_,      k * 32, D_, lane);
      fa1[k] = load_a_frag(statesB, l * B_ + 16, k * 32, D_, lane);
    }
    v8f a0 = e0, a1 = e1;
#pragma unroll
    for (int i = 0; i < 8; ++i) { a0[i] *= dec; a1[i] *= dec; }
#pragma unroll
    for (int k = 0; k < 8; ++k) {
      a0 = wmma_bf16(fa0[k], fbw[k], a0);
      a1 = wmma_bf16(fa1[k], fbw[k], a1);
    }
#pragma unroll
    for (int i = 0; i < 8; ++i) { a0[i] += bb; a1[i] += bb; }
    e0 = a0; e1 = a1;
  }
  const int rup = (lane & 16) ? 8 : 0;
#pragma unroll
  for (int i = 0; i < 8; ++i) {
    int b0 = i + rup;          // M-tile 0: batch rows 0..15
    int b1 = 16 + i + rup;     // M-tile 1: batch rows 16..31
    E[((size_t)c * B_ + b0) * H_ + ncol] = e0[i];
    E[((size_t)c * B_ + b1) * H_ + ncol] = e1[i];
  }
}

// ---------------------------------------------------------------------------
// Pass 2: serial prefix over the 128 chunk states -> carry into each chunk.
// ---------------------------------------------------------------------------
__global__ void __launch_bounds__(256)
carry_scan_kernel(const float* __restrict__ E, const float* __restrict__ A_log,
                  float* __restrict__ Carry) {
  int e = blockIdx.x * blockDim.x + threadIdx.x;   // (b,h) pair, h fastest
  if (e >= B_ * H_) return;
  int h = e & (H_ - 1);
  float dCL = __expf(A_log[h] * (float)CL_);
  float carry = 0.f;
  for (int c = 0; c < NC_; ++c) {
    Carry[(size_t)c * (B_ * H_) + e] = carry;
    carry = dCL * carry + E[(size_t)c * (B_ * H_) + e];
  }
}

// ---------------------------------------------------------------------------
// Pass 3: fused per-chunk pipeline. One block = one chunk, 512 thr = 16 waves.
// states[l] tile is double-buffered in LDS via global_load_async_to_lds_b128
// (prefetch t+1 while computing t, drained with s_wait_asynccnt).
// LDS map: [0,64K) hbuf tanh(h) bf16 | [64K,64K+256) lp | then 2x16K sbuf.
// ---------------------------------------------------------------------------
#define LP_OFF (B_ * H_ * 2)          // 65536
#define SB_OFF (LP_OFF + 256)         // 65792
#define SB_SZ  (B_ * D_ * 2)          // 16384

__global__ void __launch_bounds__(512)
fused_scan_policy_kernel(const u16* __restrict__ statesB,
                         const u16* __restrict__ BwB,
                         const u16* __restrict__ CwB,
                         const float* __restrict__ actions,
                         const float* __restrict__ A_log,
                         const float* __restrict__ B_b,
                         const float* __restrict__ C_b,
                         const float* __restrict__ log_std,
                         const float* __restrict__ Carry,
                         float* __restrict__ out) {
  extern __shared__ char smem_raw[];
  u16*   hbuf = (u16*)smem_raw;                 // [32][1024] bf16 tanh(h)
  float* lp   = (float*)(smem_raw + LP_OFF);    // [32] log-prob accum

  const int c    = blockIdx.x;
  const int tid  = threadIdx.x;
  const int w    = tid >> 5;
  const int lane = tid & 31;
  const int m     = w >> 3;       // M-tile (batch rows m*16..m*16+15)
  const int nbase = (w & 7) * 128;
  const int rup   = (lane & 16) ? 8 : 0;

  // Persistent hidden-state tiles + per-column constants
  v8f   hst[8];
  float dec[8], bb[8];
#pragma unroll
  for (int j = 0; j < 8; ++j) {
    int ncol = nbase + j * 16 + (lane & 15);
    dec[j] = __expf(A_log[ncol]);
    bb[j]  = B_b[ncol];
    v8f hv;
#pragma unroll
    for (int i = 0; i < 8; ++i) {
      int brow = m * 16 + i + rup;
      hv[i] = Carry[((size_t)c * B_ + brow) * H_ + ncol];
    }
    hst[j] = hv;
  }

  // GEMM2 epilogue constants (waves 0..7): wave -> (m2 = w>>2, action tile w&3)
  const int m2   = w >> 2;
  const int at   = w & 3;
  const int acol = at * 16 + (lane & 15);
  float inv_std = 1.f, cterm = 0.f, cbv = 0.f;
  if (w < 8) {
    float ls = log_std[acol];
    inv_std  = __expf(-ls);
    cbv      = C_b[acol];
    cterm    = -0.5f * (1.8378770664093453f + 2.f * ls);  // -(log 2pi)/2 - ls
  }

  // Prologue: async-stage states[c*CL] into staging buffer 0 (32 B/thread).
  {
    const u16* g = statesB + (size_t)(c * CL_) * B_ * D_ + tid * 16;
    u32 lo = SB_OFF + (u32)tid * 32;
    async_ld_b128(lo, g);
    async_ld_b128(lo + 16, g + 8);
  }
  wait_async0();
  __syncthreads();

  for (int t = 0; t < CL_; ++t) {
    const int l   = c * CL_ + t;
    const int cur = t & 1;
    // Current staging buffer pointer (computed dynamically: no constant
    // aggregate of LDS-derived pointers -> no static-initializer reloc).
    const u16* scur = (const u16*)(smem_raw + SB_OFF + (u32)cur * SB_SZ);

    // Kick off async staging of states[l+1] into the other buffer.
    if (t + 1 < CL_) {
      const u16* g = statesB + (size_t)(l + 1) * B_ * D_ + tid * 16;
      u32 lo = SB_OFF + (u32)(cur ^ 1) * SB_SZ + (u32)tid * 32;
      async_ld_b128(lo, g);
      async_ld_b128(lo + 16, g + 8);
    }

    // ---- GEMM1: h = dec*h + states_l @ Bw^T + bb  (A fragments from LDS)
    v16bf fa[8];
#pragma unroll
    for (int k = 0; k < 8; ++k)
      fa[k] = load_a_frag(scur, m * 16, k * 32, D_, lane);

    v8f acc[8];
#pragma unroll
    for (int j = 0; j < 8; ++j) {
      v8f a = hst[j];
#pragma unroll
      for (int i = 0; i < 8; ++i) a[i] *= dec[j];
      acc[j] = a;
    }
#pragma unroll
    for (int j = 0; j < 8; ++j) {
      v16bf fb[8];
#pragma unroll
      for (int k = 0; k < 8; ++k)
        fb[k] = load_b_frag(BwB, nbase + j * 16, k * 32, D_, lane);
#pragma unroll
      for (int k = 0; k < 8; ++k)
        acc[j] = wmma_bf16(fa[k], fb[k], acc[j]);
    }
#pragma unroll
    for (int j = 0; j < 8; ++j) {
#pragma unroll
      for (int i = 0; i < 8; ++i) acc[j][i] += bb[j];
      hst[j] = acc[j];
    }

    // ---- tanh -> bf16 -> LDS tile; zero the log-prob accumulator
#pragma unroll
    for (int j = 0; j < 8; ++j) {
      int ncol = nbase + j * 16 + (lane & 15);
#pragma unroll
      for (int i = 0; i < 8; ++i) {
        int brow = m * 16 + i + rup;
        hbuf[brow * H_ + ncol] = f2bf(fast_tanh(hst[j][i]));
      }
    }
    if (tid < B_) lp[tid] = 0.f;
    __syncthreads();

    // ---- GEMM2 (K=1024, A from LDS, batched fragment groups) + log-prob
    if (w < 8) {
      v8f macc = {};
#pragma unroll
      for (int kb = 0; kb < 4; ++kb) {
        v16bf fa2[8], fb2[8];
#pragma unroll
        for (int kk = 0; kk < 8; ++kk) {
          fa2[kk] = load_a_frag(hbuf, m2 * 16, kb * 256 + kk * 32, H_, lane);
          fb2[kk] = load_b_frag(CwB, at * 16, kb * 256 + kk * 32, H_, lane);
        }
#pragma unroll
        for (int kk = 0; kk < 8; ++kk)
          macc = wmma_bf16(fa2[kk], fb2[kk], macc);
      }
#pragma unroll
      for (int i = 0; i < 8; ++i) {
        int brow = m2 * 16 + i + rup;
        float mu = macc[i] + cbv;
        float av = actions[((size_t)l * B_ + brow) * A_ + acol];
        float dv = (av - mu) * inv_std;
        float term = cterm - 0.5f * dv * dv;
        term += __shfl_xor(term, 1, 32);   // reduce over the 16 actions
        term += __shfl_xor(term, 2, 32);   // held by lanes within each half
        term += __shfl_xor(term, 4, 32);
        term += __shfl_xor(term, 8, 32);
        if ((lane & 15) == 0) atomicAdd(&lp[brow], term);
      }
    }
    wait_async0();       // drain this iteration's async copies (all waves)
    __syncthreads();     // ... then barrier-order them for next iteration
    if (tid < B_) out[(size_t)l * B_ + tid] = lp[tid];
  }
}

// ---------------------------------------------------------------------------
// Launch
// ---------------------------------------------------------------------------
extern "C" void kernel_launch(void* const* d_in, const int* in_sizes, int n_in,
                              void* d_out, int out_size, void* d_ws, size_t ws_size,
                              hipStream_t stream) {
  (void)in_sizes; (void)n_in; (void)out_size; (void)ws_size;
  const float* states  = (const float*)d_in[0];
  const float* actions = (const float*)d_in[1];
  const float* A_log   = (const float*)d_in[2];
  const float* B_w     = (const float*)d_in[3];
  const float* B_b     = (const float*)d_in[4];
  const float* C_w     = (const float*)d_in[5];
  const float* C_b     = (const float*)d_in[6];
  const float* log_std = (const float*)d_in[7];
  float* out = (float*)d_out;

  char* ws = (char*)d_ws;
  size_t off = 0;
  u16* statesB = (u16*)(ws + off); off += (size_t)L_ * B_ * D_ * 2;    // 32 MB
  u16* BwB     = (u16*)(ws + off); off += (size_t)H_ * D_ * 2;         // 512 KB
  u16* CwB     = (u16*)(ws + off); off += (size_t)A_ * H_ * 2;         // 128 KB
  float* E     = (float*)(ws + off); off += (size_t)NC_ * B_ * H_ * 4; // 16 MB
  float* Carry = (float*)(ws + off);                                    // 16 MB

  // bf16 conversions
  cvt_bf16_kernel<<<4096, 256, 0, stream>>>(states, statesB, L_ * B_ * D_);
  cvt_bf16_kernel<<<256, 256, 0, stream>>>(B_w, BwB, H_ * D_);
  cvt_bf16_kernel<<<64, 256, 0, stream>>>(C_w, CwB, A_ * H_);

  // Pass 1: chunk-local end states
  chunk_state_kernel<<<dim3(NC_, H_ / 64), 128, 0, stream>>>(
      statesB, BwB, A_log, B_b, E);

  // Pass 2: carries
  carry_scan_kernel<<<(B_ * H_ + 255) / 256, 256, 0, stream>>>(E, A_log, Carry);

  // Pass 3: fused scan + policy head (dynamic LDS: hbuf + lp + 2x sbuf)
  const size_t smem = (size_t)SB_OFF + 2 * SB_SZ;   // 98,560 bytes
  fused_scan_policy_kernel<<<NC_, 512, smem, stream>>>(
      statesB, BwB, CwB, actions, A_log, B_b, C_b, log_std, Carry, out);
}
